// FRNet_25555055411257
// MI455X (gfx1250) — matrix operand order repaired and checked by reading, not compiled, not run on hardware
//
#include <hip/hip_runtime.h>
#include <hip/hip_bf16.h>
#include <math.h>

typedef _Float16 h16;
typedef __attribute__((ext_vector_type(16))) _Float16 v16h;
typedef __attribute__((ext_vector_type(8)))  float    v8f;

#define DEV __device__ __forceinline__

// Problem constants (from reference)
#define BB   8
#define CC   256
#define HH   128
#define WW   128
#define HWV  16384              // H*W
#define NTOK 131072             // B*H*W
#define HEADS 4
#define HD   64

DEV v8f vzero8() {
  v8f v;
#pragma unroll
  for (int i = 0; i < 8; ++i) v[i] = 0.0f;
  return v;
}

DEV v8f wmma32(v16h a, v16h b, v8f c) {
  // D = A(16x32 f16) x B(32x16 f16) + C(16x16 f32)
  return __builtin_amdgcn_wmma_f32_16x16x32_f16(
      false, a, false, b, (short)0, c, false, false);
}

// CDNA5 async global->LDS copy (16B per lane), tracked by ASYNCcnt.
// LDS address = low 32 bits of the flat shared pointer.
DEV void async_copy_b128(h16* lds_dst, const h16* gsrc) {
  unsigned lds = (unsigned)(unsigned long long)lds_dst;
  unsigned long long g = (unsigned long long)gsrc;
  asm volatile("global_load_async_to_lds_b128 %0, %1, off"
               :: "v"(lds), "v"(g) : "memory");
}
DEV void wait_async_le2() { asm volatile("s_wait_asynccnt 0x2" ::: "memory"); }
DEV void wait_async_0()   { asm volatile("s_wait_asynccnt 0x0" ::: "memory"); }

// A-fragment (16x32 f16) from LDS, row-major tile, rows `pitch` halfs apart.
// Per ISA: lane m=lane&15, hi=lane>>4; VGPR j(0..3): K=8*hi+2j,{+1};
// VGPR j(4..7): K=16+8*hi+2(j-4),{+1}.
DEV v16h lds_afrag(const h16* base, int pitch) {
  int lane = threadIdx.x & 31;
  int m = lane & 15, hi2 = lane >> 4;
  const h16* row = base + m * pitch;
  v16h a;
#pragma unroll
  for (int j = 0; j < 4; ++j) {
    a[2 * j]     = row[8 * hi2 + 2 * j];
    a[2 * j + 1] = row[8 * hi2 + 2 * j + 1];
    a[8 + 2 * j]     = row[16 + 8 * hi2 + 2 * j];
    a[8 + 2 * j + 1] = row[16 + 8 * hi2 + 2 * j + 1];
  }
  return a;
}

// B-fragment (32x16) loaded from memory where column n is K-contiguous
// (i.e. N-major storage). p must already point at (row n, k + 8*hi).
DEV v16h kfragB(const h16* p) {
  union { uint4 u[2]; v16h v; } uf;
  uf.u[0] = *(const uint4*)(p);
  uf.u[1] = *(const uint4*)(p + 16);
  return uf.v;
}

// ---------------------------------------------------------------------------
// 1) pack: x (B,C,H,W) f32 -> token-major f16 [NTOK, C]
// ---------------------------------------------------------------------------
__global__ __launch_bounds__(256)
void pack_kernel(const float* __restrict__ x, h16* __restrict__ Xt) {
  __shared__ float tile[32][33];
  int b  = blockIdx.z;
  int c0 = blockIdx.y * 32;
  int p0 = blockIdx.x * 32;
  int tx = threadIdx.x, ty = threadIdx.y;  // 32 x 8
  const float* xb = x + ((size_t)b * CC + c0) * HWV + p0;
#pragma unroll
  for (int i = 0; i < 4; ++i)
    tile[ty + 8 * i][tx] = xb[(size_t)(ty + 8 * i) * HWV + tx];
  __syncthreads();
  h16* ob = Xt + ((size_t)b * HWV + p0) * CC + c0;
#pragma unroll
  for (int i = 0; i < 4; ++i)
    ob[(size_t)(ty + 8 * i) * CC + tx] = (h16)tile[tx][ty + 8 * i];
}

// ---------------------------------------------------------------------------
// 1b) weight transpose+convert: W f32 [K][N] -> Wt f16 [N][K] (L2-resident)
// ---------------------------------------------------------------------------
__global__ __launch_bounds__(256)
void wtrans_kernel(const float* __restrict__ W, h16* __restrict__ Wt,
                   int K, int N) {
  __shared__ float tile[32][33];
  int k0 = blockIdx.x * 32;
  int n0 = blockIdx.y * 32;
  int tx = threadIdx.x, ty = threadIdx.y;  // 32 x 8
#pragma unroll
  for (int i = 0; i < 4; ++i)
    tile[ty + 8 * i][tx] = W[(size_t)(k0 + ty + 8 * i) * N + n0 + tx];
  __syncthreads();
#pragma unroll
  for (int i = 0; i < 4; ++i)
    Wt[(size_t)(n0 + ty + 8 * i) * K + k0 + tx] = (h16)tile[tx][ty + 8 * i];
}

// ---------------------------------------------------------------------------
// 2) WMMA GEMM: Out[M,N] f16 = A[M,K] f16 * Wt[N][K] f16 (+bias)
//    A staged in double-buffered LDS via async global->LDS copies;
//    B fragments loaded straight from the transposed f16 weights (L2 hits).
//    block tile 128x64, 8 waves (4x2), wave tile 32x32, BK=32
// ---------------------------------------------------------------------------
#define BM 128
#define BN 64
#define BK 32
#define AP 40   // LDS pitch (halfs) for A tile: row = 80B, 16B aligned

template <int N, int K>
__global__ __launch_bounds__(256)
void gemm_f16(const h16* __restrict__ A, const h16* __restrict__ Wt,
              const float* __restrict__ bias, h16* __restrict__ Out) {
  __shared__ __align__(16) h16 As[2][BM * AP];
  int tid = threadIdx.x;
  int n0 = blockIdx.x * BN;
  int m0 = blockIdx.y * BM;
  int w  = tid >> 5;
  int wm = (w & 3) * 32, wn = (w >> 2) * 32;
  int lane = tid & 31, ln = lane & 15, lh = lane >> 4;

  // this thread's two 16B chunks of the A tile
  int ck0 = tid * 2, ck1 = tid * 2 + 1;
  int r0 = ck0 >> 2, o0 = (ck0 & 3) * 8;
  int r1 = ck1 >> 2, o1 = (ck1 & 3) * 8;
  const h16* a0p = A + (size_t)(m0 + r0) * K + o0;
  const h16* a1p = A + (size_t)(m0 + r1) * K + o1;

  v8f acc[2][2];
  acc[0][0] = vzero8(); acc[0][1] = vzero8();
  acc[1][0] = vzero8(); acc[1][1] = vzero8();

  // preload tile 0
  async_copy_b128(&As[0][r0 * AP + o0], a0p);
  async_copy_b128(&As[0][r1 * AP + o1], a1p);

  const int NK = K / BK;
  for (int kt = 0; kt < NK; ++kt) {
    if (kt + 1 < NK) {
      // prefetch next tile into the other buffer (safe: previous readers of
      // that buffer finished before the barrier at the end of iter kt-1)
      async_copy_b128(&As[(kt + 1) & 1][r0 * AP + o0], a0p + (kt + 1) * BK);
      async_copy_b128(&As[(kt + 1) & 1][r1 * AP + o1], a1p + (kt + 1) * BK);
      wait_async_le2();   // tile kt's 2 copies (oldest) are complete
    } else {
      wait_async_0();
    }
    __syncthreads();

    const h16* At = As[kt & 1];
    v16h fa0 = lds_afrag(&At[(wm) * AP], AP);
    v16h fa1 = lds_afrag(&At[(wm + 16) * AP], AP);
    // B fragments direct from global Wt[N][K] (column n is K-contiguous)
    v16h fb0 = kfragB(Wt + (size_t)(n0 + wn + ln) * K + kt * BK + 8 * lh);
    v16h fb1 = kfragB(Wt + (size_t)(n0 + wn + 16 + ln) * K + kt * BK + 8 * lh);

    acc[0][0] = wmma32(fa0, fb0, acc[0][0]);
    acc[0][1] = wmma32(fa0, fb1, acc[0][1]);
    acc[1][0] = wmma32(fa1, fb0, acc[1][0]);
    acc[1][1] = wmma32(fa1, fb1, acc[1][1]);
    __syncthreads();
  }

  // epilogue: D layout row = 8*hi + r, col = lane&15 within each 16x16 tile
#pragma unroll
  for (int i = 0; i < 2; ++i) {
#pragma unroll
    for (int j = 0; j < 2; ++j) {
      int col = n0 + wn + j * 16 + ln;
      float bv = bias ? bias[col] : 0.0f;
      int rbase = m0 + wm + i * 16 + 8 * lh;
#pragma unroll
      for (int r = 0; r < 8; ++r)
        Out[(size_t)(rbase + r) * N + col] = (h16)(acc[i][j][r] + bv);
    }
  }
}

// ---------------------------------------------------------------------------
// 3) windowed attention: one block per 8x8 window, 8 waves.
//    qkv [NTOK,768] f16 (q|k|v), out xvec [NTOK,256] f16
// ---------------------------------------------------------------------------
__global__ __launch_bounds__(256)
void attn_kernel(const h16* __restrict__ qkv, h16* __restrict__ xvec) {
  __shared__ __align__(16) h16 vt[HEADS * HD * 72];  // V^T: [head][d][j]
  __shared__ __align__(16) h16 pbuf[8 * 16 * 72];    // per-wave P strip
  int tid = threadIdx.x;
  int wid = blockIdx.x;
  int b   = wid >> 8;
  int rem = wid & 255;
  int xw  = rem >> 4, yw = rem & 15;
  int base = b * HWV + xw * 8 * WW + yw * 8;  // token of window element (0,0)

  // stage V transposed into LDS; batch 8 loads -> 1 wait -> 8 stores
  {
    int head = tid >> 6, d = tid & 63;
    h16* vdst = &vt[(head * 64 + d) * 72];
#pragma unroll
    for (int jo = 0; jo < 64; jo += 8) {
      h16 vreg[8];
      int trow = base + ((jo >> 3) * WW);   // j&7 tokens are contiguous
#pragma unroll
      for (int ji = 0; ji < 8; ++ji)
        vreg[ji] = qkv[(size_t)(trow + ji) * 768 + 512 + tid];
#pragma unroll
      for (int ji = 0; ji < 8; ++ji) vdst[jo + ji] = vreg[ji];
    }
  }
  __syncthreads();

  int w = tid >> 5;
  int lane = tid & 31, ln = lane & 15, lh = lane >> 4;
  h16* pw = &pbuf[w * 16 * 72];
  const float scale = 0.125f;  // hd^-0.5

  for (int tsk = w; tsk < 16; tsk += 8) {
    int hh = tsk >> 2, ss = tsk & 3;

    // Q A-fragments (rows = window tokens ss*16+m), direct from global
    int qi = ss * 16 + ln;
    int tq = base + ((qi >> 3) * WW) + (qi & 7);
    const h16* qrow = qkv + (size_t)tq * 768 + hh * HD;
    v16h qf[2];
#pragma unroll
    for (int kk = 0; kk < 2; ++kk) {
      union { uint4 u[2]; v16h v; } uf;
      uf.u[0] = *(const uint4*)(qrow + kk * 32 + 8 * lh);
      uf.u[1] = *(const uint4*)(qrow + kk * 32 + 16 + 8 * lh);
      qf[kk] = uf.v;
    }

    // S = q k^T : 4 col tiles x 2 k-steps, K B-frags direct from global
    v8f st[4];
#pragma unroll
    for (int nt = 0; nt < 4; ++nt) {
      st[nt] = vzero8();
      int jn = nt * 16 + ln;
      int tk = base + ((jn >> 3) * WW) + (jn & 7);
      const h16* krow = qkv + (size_t)tk * 768 + 256 + hh * HD;
#pragma unroll
      for (int kk = 0; kk < 2; ++kk)
        st[nt] = wmma32(qf[kk], kfragB(krow + kk * 32 + 8 * lh), st[nt]);
    }

    // softmax over rows (row = 8*lh + r; cols spread over 16 lanes x 4 tiles)
#pragma unroll
    for (int r = 0; r < 8; ++r) {
      float mx = -3.4e38f;
#pragma unroll
      for (int nt = 0; nt < 4; ++nt) mx = fmaxf(mx, st[nt][r] * scale);
#pragma unroll
      for (int m = 1; m < 16; m <<= 1) mx = fmaxf(mx, __shfl_xor(mx, m, 32));
      float s = 0.0f;
#pragma unroll
      for (int nt = 0; nt < 4; ++nt) {
        float e = __expf(st[nt][r] * scale - mx);
        st[nt][r] = e;
        s += e;
      }
#pragma unroll
      for (int m = 1; m < 16; m <<= 1) s += __shfl_xor(s, m, 32);
      float inv = __frcp_rn(s);
#pragma unroll
      for (int nt = 0; nt < 4; ++nt) st[nt][r] *= inv;
    }

    // spill P (16x64) into per-wave LDS, reload as A-fragments
#pragma unroll
    for (int nt = 0; nt < 4; ++nt)
#pragma unroll
      for (int r = 0; r < 8; ++r)
        pw[(8 * lh + r) * 72 + nt * 16 + ln] = (h16)st[nt][r];
    v16h pf[2];
#pragma unroll
    for (int kk = 0; kk < 2; ++kk) pf[kk] = lds_afrag(pw + kk * 32, 72);

    // ctx = P @ V : 4 d-tiles x 2 k-steps; V B-frags from transposed LDS
#pragma unroll
    for (int dt = 0; dt < 4; ++dt) {
      v8f ct = vzero8();
      const h16* vrow = &vt[(hh * 64 + dt * 16 + ln) * 72];
#pragma unroll
      for (int kk = 0; kk < 2; ++kk)
        ct = wmma32(pf[kk], kfragB(vrow + kk * 32 + 8 * lh), ct);
#pragma unroll
      for (int r = 0; r < 8; ++r) {
        int i = ss * 16 + 8 * lh + r;
        int t = base + ((i >> 3) * WW) + (i & 7);
        xvec[(size_t)t * CC + hh * HD + dt * 16 + ln] = (h16)ct[r];
      }
    }
  }
}

// ---------------------------------------------------------------------------
// 4) instance-norm stats + normalize/GELU + bit vector + final combine
// ---------------------------------------------------------------------------
__global__ void zero_f32(float* p, int n) {
  int i = blockIdx.x * blockDim.x + threadIdx.x;
  if (i < n) p[i] = 0.0f;
}

__global__ __launch_bounds__(256)
void stats_kernel(const h16* __restrict__ hb, float* __restrict__ ssum,
                  float* __restrict__ ssq) {
  int blk = blockIdx.x;          // b*128 + ptile
  int b = blk >> 7, pt = blk & 127;
  size_t t0 = (size_t)b * HWV + pt * 128;
  int tid = threadIdx.x;
#pragma unroll
  for (int half = 0; half < 2; ++half) {
    int ch = tid + half * 256;
    float s = 0.0f, q = 0.0f;
    for (int i = 0; i < 128; ++i) {
      float v = (float)hb[(t0 + i) * 512 + ch];
      s += v;
      q += v * v;
    }
    atomicAdd(&ssum[b * 512 + ch], s);
    atomicAdd(&ssq[b * 512 + ch], q);
  }
}

__global__ void normgelu_kernel(h16* hb, const float* __restrict__ ssum,
                                const float* __restrict__ ssq) {
  const float invN = 1.0f / (float)HWV;
  size_t total = (size_t)NTOK * 512;
  size_t stride = (size_t)gridDim.x * blockDim.x;
  for (size_t idx = (size_t)blockIdx.x * blockDim.x + threadIdx.x; idx < total;
       idx += stride) {
    size_t t = idx >> 9;
    int ch = (int)(idx & 511);
    int b = (int)(t >> 14);
    float mu = ssum[b * 512 + ch] * invN;
    float var = ssq[b * 512 + ch] * invN - mu * mu;
    float rstd = __frsqrt_rn(var + 1e-5f);
    float xn = ((float)hb[idx] - mu) * rstd;
    float g = 0.5f * xn * (1.0f + erff(xn * 0.70710678118f));
    hb[idx] = (h16)g;
  }
}

__global__ __launch_bounds__(256)
void bit_kernel(const h16* __restrict__ mlp, const float* __restrict__ bw,
                const float* __restrict__ bb, float* __restrict__ bitv) {
  int gw = (blockIdx.x * 256 + threadIdx.x) >> 5;  // one wave per token
  int lane = threadIdx.x & 31;
  if (gw >= NTOK) return;
  const h16* row = mlp + (size_t)gw * CC;
  float s = 0.0f;
#pragma unroll
  for (int k = 0; k < 8; ++k) {
    int c = lane + 32 * k;
    s += (float)row[c] * bw[c];
  }
#pragma unroll
  for (int m = 1; m < 32; m <<= 1) s += __shfl_xor(s, m, 32);
  if (lane == 0) bitv[gw] = fmaxf(s + bb[0], 0.0f);
}

__global__ void combine_kernel(const float* __restrict__ x,
                               const h16* __restrict__ gvec,
                               const float* __restrict__ gbit,
                               const h16* __restrict__ wvec,
                               const float* __restrict__ wbit,
                               float* __restrict__ out) {
  size_t total = (size_t)BB * CC * HWV;
  size_t stride = (size_t)gridDim.x * blockDim.x;
  for (size_t idx = (size_t)blockIdx.x * blockDim.x + threadIdx.x; idx < total;
       idx += stride) {
    size_t b = idx >> 22;              // C*HW = 2^22
    size_t rem = idx & 4194303;
    int c = (int)(rem >> 14);
    size_t p = rem & 16383;
    size_t t = (b << 14) + p;
    float wv = (float)wvec[t * CC + c] * wbit[t];
    float sig = 1.0f / (1.0f + __expf(-wv));
    float gv = (float)gvec[t * CC + c] * gbit[t];
    out[idx] = x[idx] * sig + gv * (1.0f - sig);
  }
}

// ---------------------------------------------------------------------------
// launcher
// ---------------------------------------------------------------------------
extern "C" void kernel_launch(void* const* d_in, const int* in_sizes, int n_in,
                              void* d_out, int out_size, void* d_ws,
                              size_t ws_size, hipStream_t stream) {
  (void)in_sizes; (void)n_in; (void)out_size; (void)ws_size;
  const float* x = (const float*)d_in[0];

  char* p = (char*)d_ws;
  auto take = [&](size_t bytes) {
    char* r = p;
    p += (bytes + 255) & ~(size_t)255;
    return r;
  };
  h16* Xtok = (h16*)take((size_t)NTOK * CC * 2);        // 64 MiB
  char* region = take((size_t)NTOK * 768 * 2);          // 192 MiB (reused)
  h16* qkvb = (h16*)region;                              // qkv output
  h16* hbuf = (h16*)region;                              // fc1/norm (aliases qkv)
  h16* mlpb = (h16*)(region + (size_t)NTOK * 512 * 2);   // fc2 output
  h16* gvec = (h16*)take((size_t)NTOK * CC * 2);        // 64 MiB
  h16* wvec = (h16*)take((size_t)NTOK * CC * 2);        // 64 MiB
  float* gbit = (float*)take((size_t)NTOK * 4);
  float* wbit = (float*)take((size_t)NTOK * 4);
  float* ssum = (float*)take(4096 * 4);
  float* ssq  = (float*)take(4096 * 4);
  h16* WtQ = (h16*)take((size_t)768 * 256 * 2);          // transposed weights
  h16* Wt1 = (h16*)take((size_t)512 * 256 * 2);
  h16* Wt2 = (h16*)take((size_t)256 * 512 * 2);

  pack_kernel<<<dim3(HWV / 32, CC / 32, BB), dim3(32, 8), 0, stream>>>(x, Xtok);

  for (int br = 0; br < 2; ++br) {
    int o = (br == 0) ? 1 : 8;
    const float* Wqkv = (const float*)d_in[o + 0];
    const float* fc1w = (const float*)d_in[o + 1];
    const float* fc1b = (const float*)d_in[o + 2];
    const float* fc2w = (const float*)d_in[o + 3];
    const float* fc2b = (const float*)d_in[o + 4];
    const float* bw   = (const float*)d_in[o + 5];
    const float* bb   = (const float*)d_in[o + 6];
    h16* vec  = (br == 0) ? gvec : wvec;
    float* bv = (br == 0) ? gbit : wbit;

    // weight transposes (f32 [K][N] -> f16 [N][K])
    wtrans_kernel<<<dim3(256 / 32, 768 / 32), dim3(32, 8), 0, stream>>>(
        Wqkv, WtQ, 256, 768);
    wtrans_kernel<<<dim3(256 / 32, 512 / 32), dim3(32, 8), 0, stream>>>(
        fc1w, Wt1, 256, 512);
    wtrans_kernel<<<dim3(512 / 32, 256 / 32), dim3(32, 8), 0, stream>>>(
        fc2w, Wt2, 512, 256);

    gemm_f16<768, 256><<<dim3(768 / BN, NTOK / BM), 256, 0, stream>>>(
        Xtok, WtQ, nullptr, qkvb);
    attn_kernel<<<2048, 256, 0, stream>>>(qkvb, vec);
    gemm_f16<512, 256><<<dim3(512 / BN, NTOK / BM), 256, 0, stream>>>(
        Xtok, Wt1, fc1b, hbuf);
    zero_f32<<<32, 256, 0, stream>>>(ssum, 4096);
    zero_f32<<<32, 256, 0, stream>>>(ssq, 4096);
    stats_kernel<<<BB * 128, 256, 0, stream>>>(hbuf, ssum, ssq);
    normgelu_kernel<<<8192, 256, 0, stream>>>(hbuf, ssum, ssq);
    gemm_f16<256, 512><<<dim3(CC / BN, NTOK / BM), 256, 0, stream>>>(
        hbuf, Wt2, fc2b, mlpb);
    bit_kernel<<<NTOK / 8, 256, 0, stream>>>(mlpb, bw, bb, bv);
  }

  combine_kernel<<<8192, 256, 0, stream>>>(x, gvec, gbit, wvec, wbit,
                                           (float*)d_out);
}
